// Attention_44109314130194
// MI455X (gfx1250) — matrix operand order repaired and checked by reading, not compiled, not run on hardware
//
#include <hip/hip_runtime.h>
#include <hip/hip_bf16.h>

// ---------------------------------------------------------------------------
// Channel ("transposed") attention, MI455X / gfx1250, wave32 + WMMA bf16.
// x:[8,64,256,256] f32.  N = 65536 pixels per image.
// All GEMM stages are HBM-bound (~2.4 GB total @ 23.3 TB/s); WMMA bf16 with
// f32 accumulation gives far more than enough math throughput.
// This revision: native f32->bf16 casts (packed cvt instead of 4-op manual
// RNE) and 16x32-px macro-tiles with A-operand reuse in k1/k6/k7.
// ---------------------------------------------------------------------------

typedef __attribute__((ext_vector_type(16))) __bf16 v16bf;
typedef __attribute__((ext_vector_type(8)))  float  v8f;

#define LLN 65536LL   // pixels per image (256*256)

// float -> bf16: native cast; backend should select v_cvt_pk_bf16_f32.
static __device__ __forceinline__ __bf16 f2bf(float f) {
  return (__bf16)f;
}

// ---- WMMA operand loaders (contiguous k order K = half*16 + e, wave32) --------------------
// A (16x32, 16-bit): lanes 0-15 row M=lane, K=0..15; lanes 16-31
// row M=lane-16, K=16..31; vector elem e -> K=half*16+e.
// Memory: src[row*stride + kbase + K], contiguous.
static __device__ __forceinline__ v16bf load_A_contig(const float* __restrict__ src,
                                                      long long stride, int row,
                                                      int half, int kbase) {
  const float* p0 = src + (long long)row * stride + kbase + half * 16;
  const float* p1 = p0 + 8;
  v16bf a;
#pragma unroll
  for (int e = 0; e < 8; ++e) a[e] = f2bf(p0[e]);
#pragma unroll
  for (int e = 0; e < 8; ++e) a[8 + e] = f2bf(p1[e]);
  return a;
}

// B (32x16, 16-bit): lanes 0-15 col N=lane, K=0..15 (elem e -> K=e);
// lanes 16-31 col N=lane-16, K=16..31.  Two memory flavors:
//   strided: element(K,col) = src[K*stride + col]   (K-major tensor)
static __device__ __forceinline__ v16bf load_B_strided(const float* __restrict__ src,
                                                       long long stride, int col,
                                                       int half) {
  const float* p = src + (long long)(half * 16) * stride + col;
  v16bf b;
#pragma unroll
  for (int e = 0; e < 16; ++e) b[e] = f2bf(p[(long long)e * stride]);
  return b;
}
//   contig: element(K,col) = src[col*stride + kbase + K]  (K contiguous)
static __device__ __forceinline__ v16bf load_B_contig(const float* __restrict__ src,
                                                      long long stride, int col,
                                                      int half, int kbase) {
  const float* p = src + (long long)col * stride + kbase + half * 16;
  v16bf b;
#pragma unroll
  for (int e = 0; e < 16; ++e) b[e] = f2bf(p[e]);
  return b;
}

// D (16x16 f32): VGPR r, lanes 0-15 -> M=r, N=lane; lanes 16-31 -> M=r+8.
static __device__ __forceinline__ void store_D(float* __restrict__ dst,
                                               long long stride, int col, int half,
                                               v8f d) {
#pragma unroll
  for (int r = 0; r < 8; ++r)
    dst[(long long)(r + half * 8) * stride + col] = d[r];
}

static __device__ __forceinline__ v8f wmma_bf16(v16bf a, v16bf b, v8f c) {
  return __builtin_amdgcn_wmma_f32_16x16x32_bf16(false, a, false, b,
                                                 (short)0, c, false, false);
}

// ---------------------------------------------------------------------------
// K1: qkv 1x1 conv.  out[b,oc,n] = sum_ic W[oc,ic] * x[b,ic,n]
// per-wave 16oc x 32px macro-tile (A reused across both px tiles),
// K=64 in two bf16 WMMA K-chunks.
// tiles = 8b * 12oc * 2048px = 196608 ; 8 waves/block -> 24576 blocks.
__global__ void k_qkv_pointwise(const float* __restrict__ x,
                                const float* __restrict__ wq,
                                float* __restrict__ out) {
  int wave = threadIdx.x >> 5, lane = threadIdx.x & 31;
  int half = lane >> 4, rc = lane & 15;
  long long tile = (long long)blockIdx.x * 8 + wave;
  int px_t = (int)(tile & 2047);
  int t11  = (int)(tile >> 11);
  int oc_t = t11 % 12, b = t11 / 12;
  long long px0 = (long long)px_t * 32;
  const float* xb = x + (long long)b * 64 * LLN + px0;
  v8f acc0 = {};
  v8f acc1 = {};
#pragma unroll
  for (int kk = 0; kk < 2; ++kk) {
    int ic0 = kk * 32;
    v16bf A  = load_A_contig(wq + (long long)oc_t * 16 * 64, 64, rc, half, ic0);
    v16bf B0 = load_B_strided(xb + (long long)ic0 * LLN, LLN, rc, half);
    v16bf B1 = load_B_strided(xb + (long long)ic0 * LLN + 16, LLN, rc, half);
    acc0 = wmma_bf16(A, B0, acc0);
    acc1 = wmma_bf16(A, B1, acc1);
  }
  float* dst = out + ((long long)b * 192 + oc_t * 16) * LLN + px0;
  store_D(dst, LLN, rc, half, acc0);
  store_D(dst + 16, LLN, rc, half, acc1);
}

// ---------------------------------------------------------------------------
// K2: depthwise 3x3, pad 1.  4 px per thread, register-tiled 3x6 window.
__global__ void k_dwconv3x3(const float* __restrict__ in,
                            const float* __restrict__ wdw,
                            float* __restrict__ out) {
  int bc  = blockIdx.y;                                   // b*192 + c
  int idx = blockIdx.x * blockDim.x + threadIdx.x;        // 0..16383
  int y = idx >> 6;
  int x0 = (idx & 63) << 2;
  int c = bc % 192;
  float wv[9];
#pragma unroll
  for (int i = 0; i < 9; ++i) wv[i] = wdw[c * 9 + i];
  const float* src = in + (long long)bc * LLN;
  float v[3][6];
#pragma unroll
  for (int dy = 0; dy < 3; ++dy) {
    int yy = y + dy - 1;
#pragma unroll
    for (int dx = 0; dx < 6; ++dx) {
      int xx = x0 + dx - 1;
      bool ok = (yy >= 0) && (yy < 256) && (xx >= 0) && (xx < 256);
      v[dy][dx] = ok ? src[yy * 256 + xx] : 0.f;
    }
  }
  float* dst = out + (long long)bc * LLN + y * 256 + x0;
#pragma unroll
  for (int j = 0; j < 4; ++j) {
    float acc = 0.f;
#pragma unroll
    for (int dy = 0; dy < 3; ++dy)
#pragma unroll
      for (int dx = 0; dx < 3; ++dx)
        acc += wv[dy * 3 + dx] * v[dy][j + dx];
    dst[j] = acc;
  }
}

// ---------------------------------------------------------------------------
// K3: inverse L2 norms for q (ch 0..63) and k (ch 64..127) of qkv2.
// one block per (b, c2), c2 in [0,128).  invn[b*128+c2] = 1/max(||.||,1e-12)
__global__ void k_norms(const float* __restrict__ qkv2, float* __restrict__ invn) {
  int bc = blockIdx.x;
  int b = bc >> 7, c2 = bc & 127;
  const float4* p = (const float4*)(qkv2 + ((long long)b * 192 + c2) * LLN);
  float s = 0.f;
  for (int i = threadIdx.x; i < 16384; i += 256) {
    float4 v = p[i];
    s += v.x * v.x + v.y * v.y + v.z * v.z + v.w * v.w;
  }
  __shared__ float red[256];
  red[threadIdx.x] = s;
  __syncthreads();
  for (int off = 128; off > 0; off >>= 1) {
    if (threadIdx.x < off) red[threadIdx.x] += red[threadIdx.x + off];
    __syncthreads();
  }
  if (threadIdx.x == 0) invn[bc] = 1.f / fmaxf(sqrtf(red[0]), 1e-12f);
}

__global__ void k_zero(float* __restrict__ p, int n) {
  int i = blockIdx.x * blockDim.x + threadIdx.x;
  if (i < n) p[i] = 0.f;
}

// ---------------------------------------------------------------------------
// K4: attn[b,h,i,j] = invq[i]*invk[j] * sum_n q[i,n]*k[j,n]   (split-K + atomics)
// grid (128 ksplit, 16 bh) x 128 thr (4 waves = the four 16x16 tiles of 32x32).
__global__ void k_attn_partial(const float* __restrict__ qkv2,
                               const float* __restrict__ invn,
                               float* __restrict__ attn) {
  int wave = threadIdx.x >> 5, lane = threadIdx.x & 31;
  int half = lane >> 4, rc = lane & 15;
  int ti = wave >> 1, tj = wave & 1;
  int bh = blockIdx.y, b = bh >> 1, h = bh & 1;
  int n0base = blockIdx.x * 512;
  const float* qb = qkv2 + ((long long)b * 192 + h * 32 + ti * 16) * LLN;
  const float* kb = qkv2 + ((long long)b * 192 + 64 + h * 32 + tj * 16) * LLN;
  v8f acc = {};
  for (int it = 0; it < 16; ++it) {
    int n0 = n0base + it * 32;
    v16bf A = load_A_contig(qb, LLN, rc, half, n0);   // A: 16 q-rows x 32 n
    v16bf B = load_B_contig(kb, LLN, rc, half, n0);   // B: 32 n x 16 k-cols
    acc = wmma_bf16(A, B, acc);
  }
  int j = tj * 16 + rc;
  float invk = invn[b * 128 + 64 + h * 32 + j];
#pragma unroll
  for (int r = 0; r < 8; ++r) {
    int i = ti * 16 + half * 8 + r;
    float s = invn[b * 128 + h * 32 + i] * invk;
    atomicAdd(attn + ((long long)bh * 32 + i) * 32 + j, acc[r] * s);
  }
}

// ---------------------------------------------------------------------------
// K5: temperature scale + exact top-k (index tie-break like lax.top_k) masked
// softmax for k=16,21,24, combined with a1,a2,a3.  1 lane per row.
__global__ void k_topk_softmax(const float* __restrict__ attn,
                               const float* __restrict__ temp,
                               const float* __restrict__ a1,
                               const float* __restrict__ a2,
                               const float* __restrict__ a3,
                               float* __restrict__ attnc) {
  int bh = blockIdx.x, h = bh & 1, lane = threadIdx.x;
  float t = temp[h];
  const float* ap = attn + (long long)bh * 1024 + lane * 32;
  float row[32];
  int rank[32];
#pragma unroll
  for (int j = 0; j < 32; ++j) row[j] = ap[j] * t;
#pragma unroll
  for (int j = 0; j < 32; ++j) {
    int c = 0;
#pragma unroll
    for (int jj = 0; jj < 32; ++jj)
      c += (row[jj] > row[j]) || ((row[jj] == row[j]) && (jj < j));
    rank[j] = c;
  }
  float outv[32];
#pragma unroll
  for (int j = 0; j < 32; ++j) outv[j] = 0.f;
  const int   ks[3]   = {16, 21, 24};
  const float coef[3] = {a1[0], a2[0], a3[0]};
  for (int s3 = 0; s3 < 3; ++s3) {
    int k = ks[s3];
    float m = -3.4e38f;
#pragma unroll
    for (int j = 0; j < 32; ++j) if (rank[j] < k) m = fmaxf(m, row[j]);
    float sum = 0.f;
#pragma unroll
    for (int j = 0; j < 32; ++j) if (rank[j] < k) sum += __expf(row[j] - m);
    float c = coef[s3] / sum;
#pragma unroll
    for (int j = 0; j < 32; ++j) if (rank[j] < k) outv[j] += c * __expf(row[j] - m);
  }
  float* op = attnc + (long long)bh * 1024 + lane * 32;
#pragma unroll
  for (int j = 0; j < 32; ++j) op[j] = outv[j];
}

// ---------------------------------------------------------------------------
// K6: aout = attn_c @ v : per-wave 16ch x 32px macro-tile (A reused),
// one K=32 WMMA per px tile.
// tiles = 8b * 2h * 2ti * 2048px = 65536 -> 8192 blocks.
__global__ void k_attn_v(const float* __restrict__ qkv2,
                         const float* __restrict__ attnc,
                         float* __restrict__ aout) {
  int wave = threadIdx.x >> 5, lane = threadIdx.x & 31;
  int half = lane >> 4, rc = lane & 15;
  long long tile = (long long)blockIdx.x * 8 + wave;
  int px_t = (int)(tile & 2047);
  int ti   = (int)((tile >> 11) & 1);
  int h    = (int)((tile >> 12) & 1);
  int b    = (int)(tile >> 13);
  long long px0 = (long long)px_t * 32;
  int bh = b * 2 + h;
  v16bf A = load_A_contig(attnc + (long long)bh * 1024 + ti * 16 * 32, 32, rc, half, 0);
  const float* vb = qkv2 + ((long long)b * 192 + 128 + h * 32) * LLN + px0;
  v16bf B0 = load_B_strided(vb, LLN, rc, half);
  v16bf B1 = load_B_strided(vb + 16, LLN, rc, half);
  v8f acc0 = {};
  v8f acc1 = {};
  acc0 = wmma_bf16(A, B0, acc0);
  acc1 = wmma_bf16(A, B1, acc1);
  float* dst = aout + ((long long)b * 64 + h * 32 + ti * 16) * LLN + px0;
  store_D(dst, LLN, rc, half, acc0);
  store_D(dst + 16, LLN, rc, half, acc1);
}

// ---------------------------------------------------------------------------
// K7: proj 1x1 conv: out[b,oc,n] = sum_ic Wp[oc,ic]*aout[b,ic,n]
// per-wave 16oc x 32px macro-tile, K=64 in two WMMA K-chunks (A reused).
// tiles = 8b * 4oc * 2048px = 65536 -> 8192 blocks.
__global__ void k_proj(const float* __restrict__ aout,
                       const float* __restrict__ wp,
                       float* __restrict__ out) {
  int wave = threadIdx.x >> 5, lane = threadIdx.x & 31;
  int half = lane >> 4, rc = lane & 15;
  long long tile = (long long)blockIdx.x * 8 + wave;
  int px_t = (int)(tile & 2047);
  int oc_t = (int)((tile >> 11) & 3);
  int b    = (int)(tile >> 13);
  long long px0 = (long long)px_t * 32;
  v8f acc0 = {};
  v8f acc1 = {};
#pragma unroll
  for (int kk = 0; kk < 2; ++kk) {
    int ic0 = kk * 32;
    v16bf A  = load_A_contig(wp + (long long)oc_t * 16 * 64, 64, rc, half, ic0);
    const float* src = aout + ((long long)b * 64 + ic0) * LLN + px0;
    v16bf B0 = load_B_strided(src, LLN, rc, half);
    v16bf B1 = load_B_strided(src + 16, LLN, rc, half);
    acc0 = wmma_bf16(A, B0, acc0);
    acc1 = wmma_bf16(A, B1, acc1);
  }
  float* dst = out + ((long long)b * 64 + oc_t * 16) * LLN + px0;
  store_D(dst, LLN, rc, half, acc0);
  store_D(dst + 16, LLN, rc, half, acc1);
}

// ---------------------------------------------------------------------------
extern "C" void kernel_launch(void* const* d_in, const int* in_sizes, int n_in,
                              void* d_out, int out_size, void* d_ws, size_t ws_size,
                              hipStream_t stream) {
  const float* x     = (const float*)d_in[0];
  const float* wqkv  = (const float*)d_in[1];
  const float* wdw   = (const float*)d_in[2];
  const float* wproj = (const float*)d_in[3];
  const float* temp  = (const float*)d_in[4];
  const float* a1    = (const float*)d_in[5];
  const float* a2    = (const float*)d_in[6];
  const float* a3    = (const float*)d_in[7];
  (void)in_sizes; (void)n_in; (void)out_size; (void)ws_size;

  float* ws = (float*)d_ws;
  const long long SZ = 8LL * 192 * 65536;   // 100,663,296 floats
  float* qkv1  = ws;                        // qkv after 1x1 conv
  float* qkv2  = qkv1 + SZ;                 // qkv after depthwise 3x3
  float* invn  = qkv2 + SZ;                 // 1024   (8 x 128 inverse norms)
  float* attn  = invn + 1024;               // 16384  (8x2x32x32)
  float* attnc = attn + 16384;              // 16384  (combined softmax matrix)
  float* aout  = attnc + 16384;             // 33,554,432 (attention output)
  float* out   = (float*)d_out;

  k_qkv_pointwise<<<24576, 256, 0, stream>>>(x, wqkv, qkv1);
  k_dwconv3x3<<<dim3(64, 1536), 256, 0, stream>>>(qkv1, wdw, qkv2);
  k_norms<<<1024, 256, 0, stream>>>(qkv2, invn);
  k_zero<<<64, 256, 0, stream>>>(attn, 16384);
  k_attn_partial<<<dim3(128, 16), 128, 0, stream>>>(qkv2, invn, attn);
  k_topk_softmax<<<16, 32, 0, stream>>>(attn, temp, a1, a2, a3, attnc);
  k_attn_v<<<8192, 256, 0, stream>>>(qkv2, attnc, aout);
  k_proj<<<8192, 256, 0, stream>>>(aout, wproj, out);
}
